// MambaPlusPlus_regular_88596585382316
// MI455X (gfx1250) — matrix operand order repaired and checked by reading, not compiled, not run on hardware
//
#include <hip/hip_runtime.h>

// ---------------------------------------------------------------------------
// MI455X (gfx1250, wave32) implementation of the MambaPlusPlus block.
// All large GEMMs use v_wmma_f32_16x16x32_bf16 (bf16 inputs, f32 accumulate).
// Weights are converted once per launch to transposed bf16 [N][K] so that
// WMMA B-fragments are single contiguous 32B loads.
// ---------------------------------------------------------------------------

typedef __attribute__((ext_vector_type(16))) __bf16          v16bf;
typedef __attribute__((ext_vector_type(8)))  float           v8f;
typedef __attribute__((ext_vector_type(8)))  unsigned short  v8u;
typedef __attribute__((ext_vector_type(16))) unsigned short  v16u;

#define T_TOK 4096   // B*L tokens
#define E_DIM 1024
#define H_DIM 1024
#define F_DIM 4096   // 4*H
#define V_DIM 32000
#define L_SEQ 2048
#define B_SZ  2

__device__ __forceinline__ unsigned short f2bf(float f) {
  unsigned int u = __float_as_uint(f);
  u += 0x7FFFu + ((u >> 16) & 1u);   // round-to-nearest-even
  return (unsigned short)(u >> 16);
}

enum { EP_F32 = 0, EP_TANH_F32, EP_MULGATE_F32, EP_GELU_BF16, EP_ADDRES_BF16, EP_BF16 };

// ---------------------------------------------------------------------------
// Generic GEMM: C[M][N] = A[M][K] (bf16) * Bt[N][K]^T (bf16) + bias, + epilogue
// Wave tile 64 x (16*NI); block = 8 waves (2 x 4) -> tile 128 x (64*NI).
//   NI=2: block 128x128 (used for N=1024 GEMMs, keeps grid large)
//   NI=4: block 128x256 (used for N=4096 / N=32000 GEMMs, 1.0 load/WMMA)
// ---------------------------------------------------------------------------
template <int NI>
__global__ __launch_bounds__(256) void gemm_bf16_wmma(
    const unsigned short* __restrict__ A,    // [M][K] bf16 bits
    const unsigned short* __restrict__ Bt,   // [N][K] bf16 bits (transposed weight)
    const float* __restrict__ bias,          // [N]
    const float* __restrict__ aux,           // [M][N] (gate / residual) or null
    float* __restrict__ outF, unsigned short* __restrict__ outB,
    int M, int N, int K, int mode)
{
  const int lane = threadIdx.x & 31;
  const int wave = threadIdx.x >> 5;
  const int wm   = wave & 1;
  const int wn   = wave >> 1;
  const int l15  = lane & 15;
  const int lh   = lane >> 4;              // 0 or 1 (half-wave)
  const int bm   = blockIdx.y * 128 + wm * 64;
  const int bn   = blockIdx.x * (64 * NI) + wn * (16 * NI);

  v8f acc[4][NI];
  const v8f vzero = {0.f, 0.f, 0.f, 0.f, 0.f, 0.f, 0.f, 0.f};
#pragma unroll
  for (int i = 0; i < 4; ++i)
#pragma unroll
    for (int j = 0; j < NI; ++j) acc[i][j] = vzero;

  const int aLo = lh * 16;                 // A: K runs [aLo, aLo+16) contiguous
  for (int k0 = 0; k0 < K; k0 += 32) {
    v16bf af[4];
    const unsigned short* aps[4];
#pragma unroll
    for (int mi = 0; mi < 4; ++mi) {
      const unsigned short* ap = A + (size_t)(bm + mi * 16 + l15) * K + (k0 + aLo);
      aps[mi] = ap;
      v8u a0 = *(const v8u*)ap;            // K = aLo .. aLo+7
      v8u a1 = *(const v8u*)(ap + 8);      // K = aLo+8 .. aLo+15
      v16u av = __builtin_shufflevector(a0, a1,
          0, 1, 2, 3, 4, 5, 6, 7, 8, 9, 10, 11, 12, 13, 14, 15);
      af[mi] = __builtin_bit_cast(v16bf, av);
    }
    v16bf bfr[NI];
#pragma unroll
    for (int ni = 0; ni < NI; ++ni) {
      const unsigned short* bp = Bt + (size_t)(bn + ni * 16 + l15) * K + (k0 + lh * 16);
      v16u bv = *(const v16u*)bp;          // contiguous 16 K-values for this column
      bfr[ni] = __builtin_bit_cast(v16bf, bv);
    }
    // Prefetch streaming A two k-steps ahead (single guard -> one branch,
    // clause of global_prefetch_b8; A fragments are re-read by the 3 other
    // waves sharing this wm row-strip, so near-cache residency pays off).
    if (k0 + 64 < K) {
#pragma unroll
      for (int mi = 0; mi < 4; ++mi)
        __builtin_prefetch(aps[mi] + 64, 0, 3);
    }
#pragma unroll
    for (int mi = 0; mi < 4; ++mi)
#pragma unroll
      for (int ni = 0; ni < NI; ++ni)
        acc[mi][ni] = __builtin_amdgcn_wmma_f32_16x16x32_bf16(
            false, af[mi], false, bfr[ni], (short)0, acc[mi][ni], false, false);
  }

  // Epilogue. C/D layout: VGPR v -> row (v + lh*8), lane&15 -> col.
#pragma unroll
  for (int mi = 0; mi < 4; ++mi) {
#pragma unroll
    for (int ni = 0; ni < NI; ++ni) {
      const int col = bn + ni * 16 + l15;
      const float bcol = bias[col];
#pragma unroll
      for (int v = 0; v < 8; ++v) {
        const int row = bm + mi * 16 + lh * 8 + v;
        const size_t oi = (size_t)row * N + col;
        float val = acc[mi][ni][v] + bcol;
        switch (mode) {
          case EP_TANH_F32:    outF[oi] = tanhf(val); break;
          case EP_MULGATE_F32: outF[oi] = val * aux[oi]; break;
          case EP_GELU_BF16:
            outB[oi] = f2bf(0.5f * val * (1.f + erff(val * 0.7071067811865476f)));
            break;
          case EP_ADDRES_BF16: outB[oi] = f2bf(val + aux[oi]); break;
          case EP_BF16:        outB[oi] = f2bf(val); break;
          default:             outF[oi] = val; break;
        }
      }
    }
  }
}

// ---------------------------------------------------------------------------
// Embedding gather -> bf16 activations
// ---------------------------------------------------------------------------
__global__ __launch_bounds__(256) void embed_kernel(
    const int* __restrict__ x, const float* __restrict__ tbl,
    unsigned short* __restrict__ ebf)
{
  const int i = blockIdx.x * 256 + threadIdx.x;     // < T_TOK*E_DIM = 4M
  const int t = i >> 10, e = i & 1023;
  ebf[i] = f2bf(tbl[(size_t)x[t] * E_DIM + e]);
}

// Per-head projection weights [NH][E][HD] -> transposed bf16 [NH*HD][E]
__global__ __launch_bounds__(256) void cvt_headw_kernel(
    const float* __restrict__ src, unsigned short* __restrict__ dst)
{
  const int i = blockIdx.x * 256 + threadIdx.x;     // < 1M ; dst[n*1024 + e]
  const int n = i >> 10, e = i & 1023;
  const int h = n >> 6, d = n & 63;
  dst[i] = f2bf(src[h * 65536 + e * 64 + d]);
}

// WC [NH][HD][HD] -> block-diagonal transposed bf16 [1024][1024]
__global__ __launch_bounds__(256) void cvt_wcfull_kernel(
    const float* __restrict__ wc, unsigned short* __restrict__ dst)
{
  const int i = blockIdx.x * 256 + threadIdx.x;     // dst[n*1024 + k]
  const int n = i >> 10, k = i & 1023;
  const int hn = n >> 6, hk = k >> 6;
  const float v = (hn == hk) ? wc[hn * 4096 + (k & 63) * 64 + (n & 63)] : 0.f;
  dst[i] = f2bf(v);
}

// LDS-tiled transpose+convert: src f32 [K][N] -> dst bf16 [N][K]
__global__ __launch_bounds__(256) void cvt_transpose_kernel(
    const float* __restrict__ src, unsigned short* __restrict__ dst, int K, int N)
{
  __shared__ unsigned short tile[64][65];
  const int kt = blockIdx.y * 64, nt = blockIdx.x * 64;
  const int tx = threadIdx.x & 63, ty = threadIdx.x >> 6;  // 64 x 4
#pragma unroll
  for (int j = 0; j < 64; j += 4)
    tile[ty + j][tx] = f2bf(src[(size_t)(kt + ty + j) * N + (nt + tx)]);
  __syncthreads();
#pragma unroll
  for (int j = 0; j < 64; j += 4)
    dst[(size_t)(nt + ty + j) * K + (kt + tx)] = tile[tx][ty + j];
}

// ---------------------------------------------------------------------------
// Gated linear recurrence h_t = a_t * h_{t-1} + b_t, per (batch, channel).
// Only 2048 independent chains -> unroll so independent loads of successive
// timesteps are batched ahead of the dependent FMA chain.
// ---------------------------------------------------------------------------
__global__ __launch_bounds__(256) void scan_kernel(
    const float* __restrict__ a, const float* __restrict__ bt,
    unsigned short* __restrict__ hb)
{
  const int id = blockIdx.x * 256 + threadIdx.x;    // < B_SZ*H_DIM = 2048
  const int b = id / H_DIM, c = id % H_DIM;
  const size_t base = (size_t)b * L_SEQ * H_DIM + c;
  float h = 0.f;
#pragma unroll 8
  for (int t = 0; t < L_SEQ; ++t) {
    const size_t idx = base + (size_t)t * H_DIM;
    h = a[idx] * h + bt[idx];
    hb[idx] = f2bf(h);
  }
}

// ---------------------------------------------------------------------------
// u = z + LayerNorm(z); emits f32 (residual consumer) and bf16 (GEMM input)
// ---------------------------------------------------------------------------
__global__ __launch_bounds__(256) void layernorm_kernel(
    const float* __restrict__ z, const float* __restrict__ g,
    const float* __restrict__ bvec, float* __restrict__ u,
    unsigned short* __restrict__ ub)
{
  __shared__ float red0[8], red1[8];
  const int row = blockIdx.x;
  const float* zr = z + (size_t)row * H_DIM;
  float vals[4], s = 0.f, s2 = 0.f;
#pragma unroll
  for (int i = 0; i < 4; ++i) {
    const float v = zr[threadIdx.x + i * 256];
    vals[i] = v; s += v; s2 += v * v;
  }
#pragma unroll
  for (int off = 16; off > 0; off >>= 1) {
    s  += __shfl_down(s,  off, 32);
    s2 += __shfl_down(s2, off, 32);
  }
  const int wv = threadIdx.x >> 5, ln = threadIdx.x & 31;
  if (ln == 0) { red0[wv] = s; red1[wv] = s2; }
  __syncthreads();
  if (threadIdx.x == 0) {
    float aa = 0.f, bb2 = 0.f;
    for (int i = 0; i < 8; ++i) { aa += red0[i]; bb2 += red1[i]; }
    red0[0] = aa; red1[0] = bb2;
  }
  __syncthreads();
  const float mu  = red0[0] * (1.f / H_DIM);
  const float var = red1[0] * (1.f / H_DIM) - mu * mu;
  const float inv = rsqrtf(var + 1e-5f);
#pragma unroll
  for (int i = 0; i < 4; ++i) {
    const int c = threadIdx.x + i * 256;
    const float val = vals[i] + (vals[i] - mu) * inv * g[c] + bvec[c];
    u [(size_t)row * H_DIM + c] = val;
    ub[(size_t)row * H_DIM + c] = f2bf(val);
  }
}

// ---------------------------------------------------------------------------
extern "C" void kernel_launch(void* const* d_in, const int* in_sizes, int n_in,
                              void* d_out, int out_size, void* d_ws, size_t ws_size,
                              hipStream_t stream) {
  (void)in_sizes; (void)n_in; (void)out_size; (void)ws_size;

  const int*   x    = (const int*)  d_in[0];
  const float* etab = (const float*)d_in[1];
  const float* Wa   = (const float*)d_in[2];
  const float* ba   = (const float*)d_in[3];
  const float* Wb   = (const float*)d_in[4];
  const float* bb   = (const float*)d_in[5];
  const float* Wo   = (const float*)d_in[6];
  const float* bo   = (const float*)d_in[7];
  const float* WC   = (const float*)d_in[8];
  const float* bC   = (const float*)d_in[9];
  const float* ng   = (const float*)d_in[10];
  const float* nb   = (const float*)d_in[11];
  const float* W1   = (const float*)d_in[12];
  const float* b1   = (const float*)d_in[13];
  const float* W2   = (const float*)d_in[14];
  const float* b2   = (const float*)d_in[15];
  const float* Wp   = (const float*)d_in[16];
  const float* bp   = (const float*)d_in[17];
  const float* Wfc  = (const float*)d_in[18];
  const float* bfc  = (const float*)d_in[19];
  float* out = (float*)d_out;

  char* ws = (char*)d_ws;
  const size_t MB = 1024ull * 1024ull;
  // Workspace layout (regions reused when lifetimes are disjoint; stream order
  // serializes all kernels so reuse is safe):
  unsigned short* ebf  = (unsigned short*)(ws + 0);        // 8MB  (later: hbf, houtbf)
  unsigned short* waT  = (unsigned short*)(ws + 8  * MB);  // 2MB
  unsigned short* wbT  = (unsigned short*)(ws + 10 * MB);  // 2MB
  unsigned short* woT  = (unsigned short*)(ws + 12 * MB);  // 2MB
  unsigned short* wcT  = (unsigned short*)(ws + 14 * MB);  // 2MB
  unsigned short* w1T  = (unsigned short*)(ws + 16 * MB);  // 8MB
  unsigned short* w2T  = (unsigned short*)(ws + 24 * MB);  // 8MB
  unsigned short* wpT  = (unsigned short*)(ws + 32 * MB);  // 2MB
  unsigned short* wfcT = (unsigned short*)(ws + 34 * MB);  // 64MB
  float* abuf  = (float*)(ws + 98  * MB);                  // 16MB (later: zbuf)
  float* btbuf = (float*)(ws + 114 * MB);                  // 16MB (later: ubuf)
  float* gate  = (float*)(ws + 130 * MB);                  // 16MB (later: ubf + pbf)
  unsigned short* gbf = (unsigned short*)(ws + 146 * MB);  // 32MB, end = 178MB
  unsigned short* hbf    = ebf;
  unsigned short* houtbf = ebf;
  float* zbuf = abuf;
  float* ubuf = btbuf;
  unsigned short* ubf = (unsigned short*)(ws + 130 * MB);
  unsigned short* pbf = (unsigned short*)(ws + 138 * MB);

  // --- stage 0: gathers / weight conversion (transposed bf16) ---
  embed_kernel<<<(T_TOK * E_DIM) / 256, 256, 0, stream>>>(x, etab, ebf);
  cvt_headw_kernel<<<4096, 256, 0, stream>>>(Wa, waT);
  cvt_headw_kernel<<<4096, 256, 0, stream>>>(Wb, wbT);
  cvt_headw_kernel<<<4096, 256, 0, stream>>>(Wo, woT);
  cvt_wcfull_kernel<<<4096, 256, 0, stream>>>(WC, wcT);
  cvt_transpose_kernel<<<dim3(F_DIM / 64, H_DIM / 64), 256, 0, stream>>>(W1, w1T, H_DIM, F_DIM);
  cvt_transpose_kernel<<<dim3(H_DIM / 64, F_DIM / 64), 256, 0, stream>>>(W2, w2T, F_DIM, H_DIM);
  cvt_transpose_kernel<<<dim3(H_DIM / 64, H_DIM / 64), 256, 0, stream>>>(Wp, wpT, H_DIM, H_DIM);
  cvt_transpose_kernel<<<dim3(V_DIM / 64, H_DIM / 64), 256, 0, stream>>>(Wfc, wfcT, H_DIM, V_DIM);

  // --- stage 1: projections a = tanh(.), b, gate (N=1024 -> NI=2) ---
  gemm_bf16_wmma<2><<<dim3(H_DIM / 128, T_TOK / 128), 256, 0, stream>>>(
      ebf, waT, ba, nullptr, abuf, nullptr, T_TOK, H_DIM, E_DIM, EP_TANH_F32);
  gemm_bf16_wmma<2><<<dim3(H_DIM / 128, T_TOK / 128), 256, 0, stream>>>(
      ebf, wbT, bb, nullptr, btbuf, nullptr, T_TOK, H_DIM, E_DIM, EP_F32);
  gemm_bf16_wmma<2><<<dim3(H_DIM / 128, T_TOK / 128), 256, 0, stream>>>(
      ebf, woT, bo, nullptr, gate, nullptr, T_TOK, H_DIM, E_DIM, EP_F32);

  // --- stage 2: gated linear recurrence ---
  scan_kernel<<<(B_SZ * H_DIM) / 256, 256, 0, stream>>>(abuf, btbuf, hbf);

  // --- stage 3: z = (h @ WC + bC) * gate  (block-diagonal WC) ---
  gemm_bf16_wmma<2><<<dim3(H_DIM / 128, T_TOK / 128), 256, 0, stream>>>(
      hbf, wcT, bC, gate, zbuf, nullptr, T_TOK, H_DIM, H_DIM, EP_MULGATE_F32);

  // --- stage 4: u = z + LN(z) ---
  layernorm_kernel<<<T_TOK, 256, 0, stream>>>(zbuf, ng, nb, ubuf, ubf);

  // --- stage 5: FFN (N=4096 -> NI=4 wide tile) ---
  gemm_bf16_wmma<4><<<dim3(F_DIM / 256, T_TOK / 128), 256, 0, stream>>>(
      ubf, w1T, b1, nullptr, nullptr, gbf, T_TOK, F_DIM, H_DIM, EP_GELU_BF16);
  gemm_bf16_wmma<2><<<dim3(H_DIM / 128, T_TOK / 128), 256, 0, stream>>>(
      gbf, w2T, b2, ubuf, nullptr, houtbf, T_TOK, H_DIM, F_DIM, EP_ADDRES_BF16);

  // --- stage 6: head (Wfc N=32000 -> NI=4 wide tile, 125x32 = 4000 blocks) ---
  gemm_bf16_wmma<2><<<dim3(H_DIM / 128, T_TOK / 128), 256, 0, stream>>>(
      houtbf, wpT, bp, nullptr, nullptr, pbf, T_TOK, H_DIM, H_DIM, EP_BF16);
  gemm_bf16_wmma<4><<<dim3(V_DIM / 256, T_TOK / 128), 256, 0, stream>>>(
      pbf, wfcT, bfc, nullptr, out, nullptr, T_TOK, V_DIM, H_DIM, EP_F32);
}
